// InfPamAtt_85237920956612
// MI455X (gfx1250) — matrix operand order repaired and checked by reading, not compiled, not run on hardware
//
#include <hip/hip_runtime.h>
#include <hip/hip_fp16.h>

// ---------------------------------------------------------------------------
// InfPamAtt on MI455X (gfx1250): flash-attention with CDNA5 WMMA.
//   B=4, C=64, INF=64, DQK=8, N=H*W=4096.
// Phase 1: QKV projection. Q,K -> f32 [B][N][8]; V -> f16 [B][C][N].
// Phase 2: streaming softmax-attention, wave32 WMMA, 64-column chunks:
//   S tiles  : v_wmma_f32_16x16x4_f32   (exact f32, K=8 = 2x K=4)
//   O accum  : v_wmma_f32_16x16x32_f16  (P,V f16; f32 accumulate)
//   single-instruction DPP16 row reductions (v_max_num_f32_dpp/v_add_f32_dpp);
//   double-buffered async-to-LDS V staging (1 barrier/chunk, latency hidden);
//   LDS-transposed coalesced epilogue.
// ---------------------------------------------------------------------------

typedef __attribute__((ext_vector_type(2)))  float        v2f;
typedef __attribute__((ext_vector_type(8)))  float        v8f;
typedef __attribute__((ext_vector_type(16))) _Float16     v16h;
typedef __attribute__((ext_vector_type(8)))  unsigned int v8u;
typedef __attribute__((ext_vector_type(4)))  int          v4i;

#define BB    4
#define CC    64
#define INFC  64
#define DQ    8
#define NN    4096
#define CH    64                 // columns per streamed chunk
#define NCHK  (NN / CH)
#define LOG2E 1.44269504088896340f

#define GLOBAL_AS __attribute__((address_space(1)))
#define LDS_AS    __attribute__((address_space(3)))

#if defined(__has_builtin)
#  if __has_builtin(__builtin_amdgcn_global_load_async_to_lds_b128)
#    define HAVE_ASYNC_LDS 1
#  endif
#endif
#ifndef HAVE_ASYNC_LDS
#  define HAVE_ASYNC_LDS 0
#endif

// ---- DPP16 16-lane reductions, one VALU op per step -----------------------
// dst = op(dpp_perm(src0), src1); all ops stay inside each 16-lane half.
#define RED_STEP_MAX(v, CTRL)                                              \
    do { float _o;                                                         \
        asm("v_max_num_f32_dpp %0, %1, %2 " CTRL                           \
            " row_mask:0xf bank_mask:0xf"                                  \
            : "=v"(_o) : "v"(v), "v"(v));                                  \
        (v) = _o; } while (0)
#define RED_STEP_ADD(v, CTRL)                                              \
    do { float _o;                                                         \
        asm("v_add_f32_dpp %0, %1, %2 " CTRL                               \
            " row_mask:0xf bank_mask:0xf"                                  \
            : "=v"(_o) : "v"(v), "v"(v));                                  \
        (v) = _o; } while (0)

__device__ __forceinline__ float red16Max(float v) {
    RED_STEP_MAX(v, "quad_perm:[1,0,3,2]");   // xor 1
    RED_STEP_MAX(v, "quad_perm:[2,3,0,1]");   // xor 2
    RED_STEP_MAX(v, "row_half_mirror");       // pair quads within 8
    RED_STEP_MAX(v, "row_mirror");            // pair 8s within 16
    return v;
}
__device__ __forceinline__ float red16Sum(float v) {
    RED_STEP_ADD(v, "quad_perm:[1,0,3,2]");
    RED_STEP_ADD(v, "quad_perm:[2,3,0,1]");
    RED_STEP_ADD(v, "row_half_mirror");
    RED_STEP_ADD(v, "row_mirror");
    return v;
}

// ---------------------------------------------------------------------------
// Phase 1: per-pixel channel matmuls. One thread per pixel.
// ---------------------------------------------------------------------------
__global__ __launch_bounds__(256) void qkv_proj_kernel(
    const float* __restrict__ x, const float* __restrict__ inf_,
    const float* __restrict__ Wq, const float* __restrict__ bq,
    const float* __restrict__ Wk, const float* __restrict__ bk,
    const float* __restrict__ Wv, const float* __restrict__ bv,
    float* __restrict__ Qo, float* __restrict__ Ko, _Float16* __restrict__ Vt)
{
    const int gid = blockIdx.x * 256 + threadIdx.x;   // 0 .. B*N-1
    const int b   = gid >> 12;
    const int pix = gid & (NN - 1);

    float qa[DQ], ka[DQ], va[CC];
#pragma unroll
    for (int d = 0; d < DQ; ++d) { qa[d] = bq[d]; ka[d] = bk[d]; }
#pragma unroll
    for (int c = 0; c < CC; ++c) va[c] = bv[c];

    const float* infp = inf_ + (size_t)b * INFC * NN + pix;
    const float* xp   = x    + (size_t)b * CC   * NN + pix;

    for (int ci = 0; ci < INFC; ++ci) {
        const float iv = infp[(size_t)ci * NN];   // coalesced across lanes
        const float xv = xp[(size_t)ci * NN];
#pragma unroll
        for (int d = 0; d < DQ; ++d) {
            qa[d] = fmaf(Wq[d * INFC + ci], iv, qa[d]);
            ka[d] = fmaf(Wk[d * INFC + ci], iv, ka[d]);
        }
#pragma unroll
        for (int c = 0; c < CC; ++c)
            va[c] = fmaf(Wv[c * CC + ci], xv, va[c]);
    }

    float* qrow = Qo + ((size_t)b * NN + pix) * DQ;
    float* krow = Ko + ((size_t)b * NN + pix) * DQ;
#pragma unroll
    for (int d = 0; d < DQ; ++d) { qrow[d] = qa[d]; krow[d] = ka[d]; }
#pragma unroll
    for (int c = 0; c < CC; ++c)
        Vt[((size_t)b * CC + c) * NN + pix] = (_Float16)va[c];   // coalesced per c
}

// ---------------------------------------------------------------------------
// Phase 2: flash attention. Block = 256 threads = 8 wave32s; each wave owns a
// 16-row M tile; the block shares double-buffered K (64x8 f32) and V (64x64
// f16) LDS chunks, staged one chunk ahead.
// ---------------------------------------------------------------------------
__global__ __launch_bounds__(256) void flash_attn_kernel(
    const float* __restrict__ Qg, const float* __restrict__ Kg,
    const _Float16* __restrict__ Vt, const float* __restrict__ x,
    const float* __restrict__ gamma, float* __restrict__ out)
{
    __shared__ float        ksm[2][CH * DQ];        // [buf][n][d]          4 KB
    __shared__ unsigned int vsm[2][CC * (CH / 2)];  // [buf][c][32 pairs]  16 KB
    __shared__ _Float16     psm[8][16 * CH];        // per-wave scratch    16 KB

    const int tid    = threadIdx.x;
    const int wave   = tid >> 5;
    const int lane   = tid & 31;
    const int lo     = lane & 15;         // column / M-row selector in half
    const int hiHalf = lane >> 4;         // 0: lanes 0-15, 1: lanes 16-31
    const int b      = blockIdx.x >> 5;   // 32 blocks per batch
    const int m0     = ((blockIdx.x & 31) * 8 + wave) * 16;  // wave's M tile

    // --- Q A-operand, f32 A 16x4 layout (ISA 7.12.2), K=0..7 as two halves --
    const float* qrow = Qg + ((size_t)b * NN + (m0 + lo)) * DQ;
    const v2f A01 = { qrow[0 + 2 * hiHalf], qrow[1 + 2 * hiHalf] };
    const v2f A23 = { qrow[4 + 2 * hiHalf], qrow[5 + 2 * hiHalf] };

    v8f Oacc[4];                           // 16x64 f32 accumulator, 4 N tiles
#pragma unroll
    for (int t = 0; t < 4; ++t) Oacc[t] = (v8f)0.0f;
    float mrow[8], lrow[8];
#pragma unroll
    for (int g = 0; g < 8; ++g) { mrow[g] = -3.0e38f; lrow[g] = 0.0f; }
    const float gm = gamma[0];

    // per-thread staging coordinates (uniform across chunks)
    const int vc  = tid >> 2;              // V row (channel) this thread stages
    const int grp = tid & 3;               // 16-half group within the row

    auto stageK = [&](int n0, int buf) {
        const float2 kk =
            *(const float2*)(Kg + ((size_t)b * NN + n0) * DQ + tid * 2);
        *(float2*)&ksm[buf][tid * 2] = kk;
    };
    auto stageV = [&](int n0, int buf) {
        const _Float16* vg = Vt + ((size_t)b * CC + vc) * NN + n0 + grp * 16;
        unsigned int*  dst = &vsm[buf][vc * 32 + grp * 8];
#if HAVE_ASYNC_LDS
        __builtin_amdgcn_global_load_async_to_lds_b128(
            (GLOBAL_AS v4i*)vg, (LDS_AS v4i*)dst, 0, 0);
        __builtin_amdgcn_global_load_async_to_lds_b128(
            (GLOBAL_AS v4i*)(vg + 8), (LDS_AS v4i*)(dst + 4), 0, 0);
#else
        const uint4 v0 = *(const uint4*)vg;
        const uint4 v1 = *(const uint4*)(vg + 8);
        *(uint4*)dst       = v0;
        *(uint4*)(dst + 4) = v1;
#endif
    };

    // prologue: stage chunk 0 into buffer 0
    stageK(0, 0);
    stageV(0, 0);

    for (int i = 0; i < NCHK; ++i) {
        const int buf = i & 1;
#if HAVE_ASYNC_LDS
        asm volatile("s_wait_asynccnt 0" ::: "memory");  // this buf's V landed
#endif
        __syncthreads();   // all waves' staging visible; frees buf^1 for reuse

        // issue next chunk's staging now; its latency hides under compute
        if (i + 1 < NCHK) {
            stageK((i + 1) * CH, buf ^ 1);
            stageV((i + 1) * CH, buf ^ 1);
        }

        const float* kbase = &ksm[buf][0];
        const uint4* vw4   = (const uint4*)&vsm[buf][0];  // row stride 8 uint4

        // ---- S = Q*K^T : four 16x16 tiles (K-dim 8 = 2x WMMA K=4 each) ----
        // B f32 4x16 layout: lane = N col; VGPR0: d=0 (lo)/d=2 (hi), VGPR1:
        // d=1/d=3 (then +4 for the second WMMA).
        v8f S[4];
#pragma unroll
        for (int t = 0; t < 4; ++t) {
            const float* kr = kbase + (t * 16 + lo) * DQ;
            const v2f B01 = { kr[0 + 2 * hiHalf], kr[1 + 2 * hiHalf] };
            const v2f B23 = { kr[4 + 2 * hiHalf], kr[5 + 2 * hiHalf] };
            v8f acc = (v8f)0.0f;
            acc = __builtin_amdgcn_wmma_f32_16x16x4_f32(false, A01, false, B01,
                                                        (short)0, acc, false, false);
            acc = __builtin_amdgcn_wmma_f32_16x16x4_f32(false, A23, false, B23,
                                                        (short)0, acc, false, false);
            S[t] = acc;
        }

        // ---- online softmax across this 64-col chunk (DPP16 reductions) ----
        float scale[8];
#pragma unroll
        for (int g = 0; g < 8; ++g) {
            float mx = fmaxf(fmaxf(S[0][g], S[1][g]), fmaxf(S[2][g], S[3][g]));
            mx = red16Max(mx);
            const float mnew = fmaxf(mrow[g], mx);
            scale[g] = __builtin_amdgcn_exp2f((mrow[g] - mnew) * LOG2E);
            mrow[g]  = mnew;
            const int ml = g + 8 * hiHalf;
            float rs = 0.0f;
#pragma unroll
            for (int t = 0; t < 4; ++t) {
                const float p = __builtin_amdgcn_exp2f((S[t][g] - mnew) * LOG2E);
                rs += p;
                psm[wave][ml * CH + t * 16 + lo] = (_Float16)p;  // [m][64] f16
            }
            lrow[g] = lrow[g] * scale[g] + red16Sum(rs);
        }
#pragma unroll
        for (int t = 0; t < 4; ++t)
#pragma unroll
            for (int g = 0; g < 8; ++g) Oacc[t][g] *= scale[g];

        // in-wave ordering of the LDS P transpose (writes above, reads below)
        asm volatile("s_wait_dscnt 0" ::: "memory");

        // ---- P as two f16 A 16x32 operands (K=0..31, 32..63), b128 loads ----
        // A layout: lane M=lo, K base 8*hiHalf; VGPR j<4: K=2j+base, j>=4: +16
        const uint4* pw4 = (const uint4*)&psm[wave][0];   // row stride 8 uint4
        v16h PA[2];
#pragma unroll
        for (int kc = 0; kc < 2; ++kc) {
            const uint4 u0 = pw4[lo * 8 + kc * 4 + hiHalf];
            const uint4 u1 = pw4[lo * 8 + kc * 4 + 2 + hiHalf];
            v8u pu;
            pu[0] = u0.x; pu[1] = u0.y; pu[2] = u0.z; pu[3] = u0.w;
            pu[4] = u1.x; pu[5] = u1.y; pu[6] = u1.z; pu[7] = u1.w;
            PA[kc] = __builtin_bit_cast(v16h, pu);
        }

        // ---- O += P*V : 4 column tiles x 2 K-chunks of f16 K=32 WMMA ----
        // B f16 32x16 layout: lane = N col (c); VGPR g<4 packs K=8*hiHalf+2g,+1;
        // g>=4: K=8+8*hiHalf+2g,+1 (same order as PA); b128 loads from vsm.
#pragma unroll
        for (int t = 0; t < 4; ++t) {
            const int c = t * 16 + lo;
#pragma unroll
            for (int kc = 0; kc < 2; ++kc) {
                const int  base = c * 8 + kc * 4 + hiHalf;
                const uint4 u0 = vw4[base];
                const uint4 u1 = vw4[base + 2];
                v8u vu;
                vu[0] = u0.x; vu[1] = u0.y; vu[2] = u0.z; vu[3] = u0.w;
                vu[4] = u1.x; vu[5] = u1.y; vu[6] = u1.z; vu[7] = u1.w;
                const v16h VB = __builtin_bit_cast(v16h, vu);
                Oacc[t] = __builtin_amdgcn_wmma_f32_16x16x32_f16(
                    false, PA[kc], false, VB, (short)0, Oacc[t], false, false);
            }
        }
    }

    // ---- epilogue: out[b][c][m] = gamma*O[m][c]/l[m] + x[b][c][m] ----------
    // Transpose each 16x16 O tile through per-wave LDS so global accesses run
    // contiguously along m (b128, coalesced). psm reused as float scratch:
    // [0..255] tile data ([c_local][m_local]), [256..271] per-row gamma/l.
    float* ep = (float*)&psm[wave][0];
#pragma unroll
    for (int g = 0; g < 8; ++g)           // all 16 lanes of a half store the
        ep[256 + g + 8 * hiHalf] = gm / lrow[g];   // same value: benign race
    asm volatile("s_wait_dscnt 0" ::: "memory");

    const int cl = lane >> 1;             // local c row handled by this lane
    const int ms = (lane & 1) * 8;        // m segment base (0 or 8)
    float rsc[8];
#pragma unroll
    for (int j = 0; j < 8; ++j) rsc[j] = ep[256 + ms + j];

#pragma unroll
    for (int t = 0; t < 4; ++t) {
        // scatter O tile t: ep[c_local*16 + m_local]
#pragma unroll
        for (int g = 0; g < 8; ++g)
            ep[lo * 16 + g + 8 * hiHalf] = Oacc[t][g];
        asm volatile("s_wait_dscnt 0" ::: "memory");

        const float4 o0 = *(const float4*)&ep[cl * 16 + ms];
        const float4 o1 = *(const float4*)&ep[cl * 16 + ms + 4];
        const size_t base = ((size_t)b * CC + t * 16 + cl) * NN + m0 + ms;
        const float4 x0 = *(const float4*)&x[base];
        const float4 x1 = *(const float4*)&x[base + 4];
        float4 r0, r1;
        r0.x = o0.x * rsc[0] + x0.x;  r0.y = o0.y * rsc[1] + x0.y;
        r0.z = o0.z * rsc[2] + x0.z;  r0.w = o0.w * rsc[3] + x0.w;
        r1.x = o1.x * rsc[4] + x1.x;  r1.y = o1.y * rsc[5] + x1.y;
        r1.z = o1.z * rsc[6] + x1.z;  r1.w = o1.w * rsc[7] + x1.w;
        *(float4*)&out[base]     = r0;
        *(float4*)&out[base + 4] = r1;
        // per-wave LDS ops are in order: next tile's stores queue behind the
        // loads above, so no extra wait is required before overwriting ep.
    }
}

// ---------------------------------------------------------------------------
extern "C" void kernel_launch(void* const* d_in, const int* in_sizes, int n_in,
                              void* d_out, int out_size, void* d_ws, size_t ws_size,
                              hipStream_t stream)
{
    const float* x     = (const float*)d_in[0];
    const float* inf_  = (const float*)d_in[1];
    const float* Wq    = (const float*)d_in[2];
    const float* bq    = (const float*)d_in[3];
    const float* Wk    = (const float*)d_in[4];
    const float* bk    = (const float*)d_in[5];
    const float* Wv    = (const float*)d_in[6];
    const float* bv    = (const float*)d_in[7];
    const float* gamma = (const float*)d_in[8];
    float*       out   = (float*)d_out;

    // workspace: Q (512KB f32) | K (512KB f32) | Vt (2MB f16)  -> 3MB total
    char*     ws = (char*)d_ws;
    float*    Qo = (float*)ws;
    float*    Ko = (float*)(ws + (size_t)512 * 1024);
    _Float16* Vt = (_Float16*)(ws + (size_t)1024 * 1024);

    qkv_proj_kernel<<<(BB * NN) / 256, 256, 0, stream>>>(
        x, inf_, Wq, bq, Wk, bk, Wv, bv, Qo, Ko, Vt);

    // B * (N / (8 waves * 16 rows)) = 4 * 32 = 128 blocks
    flash_attn_kernel<<<BB * (NN / 128), 256, 0, stream>>>(
        Qo, Ko, Vt, x, gamma, out);
}
